// Top2Gating_80839874445609
// MI455X (gfx1250) — compile-verified, matrix operand order checked by reading, not run on hardware
//
#include <hip/hip_runtime.h>
#include <hip/hip_bf16.h>

// ---- problem constants -----------------------------------------------------
#define BB 4
#define NN 4096
#define DD 1024
#define EE 64
#define CC 80                         // expert capacity
#define TT (BB * NN)                  // 16384 tokens
#define SZ ((size_t)TT * EE * CC)     // 83,886,080 floats per output tensor

typedef __attribute__((ext_vector_type(16))) _Float16 v16h;
typedef __attribute__((ext_vector_type(8)))  float    v8f;
typedef __attribute__((ext_vector_type(4)))  float    v4f;   // native vector for NT stores

// ---------------------------------------------------------------------------
// Kernel 0: pre-swizzle w_gating (f32 [1024,64]) into f16 WMMA B-fragment
// layout: wfrag[((kb*4 + t)*32 + lane)*16 + j]
//   k = kb*32 + (lane>>4)*8 + (j&7) + (j>=8 ? 16 : 0), col = t*16 + (lane&15)
// (mirrors the 16-bit 16x32 A-matrix layout in CDNA5 ISA)
// ---------------------------------------------------------------------------
__global__ __launch_bounds__(256) void wprep_kernel(const float* __restrict__ w,
                                                    _Float16* __restrict__ wfrag) {
  int i = blockIdx.x * 256 + threadIdx.x;      // 0 .. 65535
  int j  = i & 15;
  int l  = (i >> 4) & 31;
  int t  = (i >> 9) & 3;
  int kb = i >> 11;
  int k   = kb * 32 + ((l >> 4) * 8) + (j & 7) + ((j >= 8) ? 16 : 0);
  int col = t * 16 + (l & 15);
  wfrag[i] = (_Float16)w[(size_t)k * EE + col];
}

// ---------------------------------------------------------------------------
// Kernel 1: zero the small accumulators in workspace
// ---------------------------------------------------------------------------
__global__ __launch_bounds__(256) void init_kernel(float* proxy, int* cnt1, float* lse_sum) {
  int i = threadIdx.x;
  if (i < BB * EE) { proxy[i] = 0.0f; cnt1[i] = 0; }
  if (i == 0) lse_sum[0] = 0.0f;
}

// ---------------------------------------------------------------------------
// Kernel 2: WMMA router GEMM + softmax + top-2 + gate normalization + stats.
// Block = 128 threads (4 waves). Each wave computes a 16x64 logit tile.
// ---------------------------------------------------------------------------
__global__ __launch_bounds__(128) void gemm_top2_kernel(
    const float*    __restrict__ x,       // [TT, DD]
    const _Float16* __restrict__ wfrag,   // swizzled B fragments
    int*   __restrict__ idx1, int*   __restrict__ idx2,
    float* __restrict__ gate1, float* __restrict__ gate2,
    float* __restrict__ proxy,            // [BB*EE] f32 (atomic)
    float* __restrict__ lse_sum)          // [1] f32 (atomic)
{
  __shared__ float logits[64][EE + 1];    // padded stride: no bank conflicts
  __shared__ float lse_l[64];

  const int tid  = threadIdx.x;
  const int wave = tid >> 5;
  const int lane = tid & 31;
  const int m0   = blockIdx.x * 64 + wave * 16;

  // A fragment: lane groups split K per ISA layout
  const int row  = m0 + (lane & 15);
  const int koff = (lane >> 4) * 8;
  const float* __restrict__ ap = x + (size_t)row * DD;

  v8f c0 = {}, c1 = {}, c2 = {}, c3 = {};

  for (int kb = 0; kb < 32; ++kb) {
    const int k0 = kb * 32;
    // prefetch A tile 4 k-steps ahead (emits global_prefetch_b8)
    __builtin_prefetch(ap + k0 + 128, 0, 1);

    // ---- load + convert A fragment (f32 -> f16), 16x32 tile ----
    v4f q0 = *(const v4f*)(ap + k0 + koff);
    v4f q1 = *(const v4f*)(ap + k0 + koff + 4);
    v4f q2 = *(const v4f*)(ap + k0 + koff + 16);
    v4f q3 = *(const v4f*)(ap + k0 + koff + 20);
    v16h a;
    a[0]  = (_Float16)q0.x; a[1]  = (_Float16)q0.y; a[2]  = (_Float16)q0.z; a[3]  = (_Float16)q0.w;
    a[4]  = (_Float16)q1.x; a[5]  = (_Float16)q1.y; a[6]  = (_Float16)q1.z; a[7]  = (_Float16)q1.w;
    a[8]  = (_Float16)q2.x; a[9]  = (_Float16)q2.y; a[10] = (_Float16)q2.z; a[11] = (_Float16)q2.w;
    a[12] = (_Float16)q3.x; a[13] = (_Float16)q3.y; a[14] = (_Float16)q3.z; a[15] = (_Float16)q3.w;

    // ---- B fragments: pre-swizzled, 32B contiguous per lane (2x b128) ----
    const v16h b0 = *(const v16h*)(wfrag + (((size_t)(kb * 4 + 0) * 32) + lane) * 16);
    const v16h b1 = *(const v16h*)(wfrag + (((size_t)(kb * 4 + 1) * 32) + lane) * 16);
    const v16h b2 = *(const v16h*)(wfrag + (((size_t)(kb * 4 + 2) * 32) + lane) * 16);
    const v16h b3 = *(const v16h*)(wfrag + (((size_t)(kb * 4 + 3) * 32) + lane) * 16);

    c0 = __builtin_amdgcn_wmma_f32_16x16x32_f16(false, a, false, b0, (short)0, c0, false, false);
    c1 = __builtin_amdgcn_wmma_f32_16x16x32_f16(false, a, false, b1, (short)0, c1, false, false);
    c2 = __builtin_amdgcn_wmma_f32_16x16x32_f16(false, a, false, b2, (short)0, c2, false, false);
    c3 = __builtin_amdgcn_wmma_f32_16x16x32_f16(false, a, false, b3, (short)0, c3, false, false);
  }

  // ---- spill C tiles to LDS (C layout: lane 0-15 -> M=r, lane 16-31 -> M=8+r) ----
  {
    const int mbase = (lane >> 4) * 8;
    const int ncol  = lane & 15;
#pragma unroll
    for (int r = 0; r < 8; ++r) {
      logits[wave * 16 + mbase + r][ncol +  0] = c0[r];
      logits[wave * 16 + mbase + r][ncol + 16] = c1[r];
      logits[wave * 16 + mbase + r][ncol + 32] = c2[r];
      logits[wave * 16 + mbase + r][ncol + 48] = c3[r];
    }
  }
  __syncthreads();

  // ---- per-token: top-2 + logsumexp + normalized gates (threads 0..63) ----
  if (tid < 64) {
    const int gtok = blockIdx.x * 64 + tid;
    float m1 = -3.4e38f, m2 = -3.4e38f;
    int i1 = 0, i2 = 0;
#pragma unroll 4
    for (int e = 0; e < EE; ++e) {
      float v = logits[tid][e];
      if (v > m1)      { m2 = m1; i2 = i1; m1 = v; i1 = e; }
      else if (v > m2) { m2 = v; i2 = e; }
    }
    float s = 0.0f;
#pragma unroll 4
    for (int e = 0; e < EE; ++e) s += __expf(logits[tid][e] - m1);
    const float lse = m1 + __logf(s);
    lse_l[tid] = lse;

    const float g1 = __expf(m1 - lse);
    const float g2 = __expf(m2 - lse);
    const float inv = 1.0f / (g1 + g2 + 1e-9f);
    idx1[gtok] = i1;  idx2[gtok] = i2;
    gate1[gtok] = g1 * inv;
    gate2[gtok] = g2 * inv;
  }
  __syncthreads();

  // ---- density_1_proxy partial: thread e sums raw_gates[:, e] over 64 rows ----
  if (tid < 64) {
    const int bidx = (blockIdx.x * 64) >> 12;   // batch of this token tile
    float ps = 0.0f;
#pragma unroll 4
    for (int r = 0; r < 64; ++r) ps += __expf(logits[r][tid] - lse_l[r]);
    atomicAdd(&proxy[bidx * EE + tid], ps);
  }
  if (tid == 0) {
    float s = 0.0f;
    for (int r = 0; r < 64; ++r) s += lse_l[r];
    atomicAdd(lse_sum, s);
  }
}

// ---------------------------------------------------------------------------
// Kernel 3: deterministic per-batch capacity scan. One block per batch,
// one thread per expert; token chunks broadcast through LDS.
// ---------------------------------------------------------------------------
__global__ __launch_bounds__(64) void scan_kernel(
    const int* __restrict__ idx1, const int* __restrict__ idx2,
    int* __restrict__ pos1, int* __restrict__ pos2, int* __restrict__ cnt1)
{
  const int b = blockIdx.x;
  const int e = threadIdx.x;
  __shared__ int   buf[256];
  __shared__ float m1cnt[EE];
  const int base = b * NN;

  int cnt = 0, kept = 0;
  for (int c0 = 0; c0 < NN; c0 += 256) {
    for (int i = e; i < 256; i += 64) buf[i] = idx1[base + c0 + i];
    __syncthreads();
    for (int i = 0; i < 256; ++i) {
      if (buf[i] == e) {
        int p = cnt++;
        pos1[base + c0 + i] = (p < CC) ? p : -1;
        kept += (p < CC);
      }
    }
    __syncthreads();
  }
  cnt1[b * EE + e] = cnt;                       // pre-capacity count -> density_1
  m1cnt[e] = (float)kept * (1.0f / (float)NN);  // reference quirk: MEAN offset
  __syncthreads();

  const float off = m1cnt[e];
  int cnt2 = 0;
  for (int c0 = 0; c0 < NN; c0 += 256) {
    for (int i = e; i < 256; i += 64) buf[i] = idx2[base + c0 + i];
    __syncthreads();
    for (int i = 0; i < 256; ++i) {
      if (buf[i] == e) {
        float pf = (float)cnt2 + off;           // fractional; floor(pf)==cnt2
        pos2[base + c0 + i] = (pf < (float)CC) ? cnt2 : -1;
        ++cnt2;
      }
    }
    __syncthreads();
  }
}

// ---------------------------------------------------------------------------
// Kernel 4: streaming zero-fill of both dense output tensors (NT stores)
// ---------------------------------------------------------------------------
__global__ __launch_bounds__(256) void zero_kernel(v4f* __restrict__ p, long long n4) {
  long long i      = (long long)blockIdx.x * blockDim.x + threadIdx.x;
  long long stride = (long long)gridDim.x * blockDim.x;
  v4f z = {0.0f, 0.0f, 0.0f, 0.0f};
  for (; i < n4; i += stride) __builtin_nontemporal_store(z, p + i);
}

// ---------------------------------------------------------------------------
// Kernel 5: sparse scatter (<=2 nonzeros per token)
// ---------------------------------------------------------------------------
__global__ __launch_bounds__(256) void scatter_kernel(
    const int* __restrict__ idx1, const int* __restrict__ idx2,
    const int* __restrict__ pos1, const int* __restrict__ pos2,
    const float* __restrict__ gate1, const float* __restrict__ gate2,
    float* __restrict__ dispatch, float* __restrict__ combine)
{
  const int t = blockIdx.x * 256 + threadIdx.x;   // 0 .. TT-1
  const size_t base = (size_t)t * (EE * CC);
  int p1 = pos1[t];
  if (p1 >= 0) {
    float g = gate1[t];
    size_t o = base + (size_t)idx1[t] * CC + p1;
    combine[o]  = g;
    dispatch[o] = (g != 0.0f) ? 1.0f : 0.0f;
  }
  int p2 = pos2[t];
  if (p2 >= 0) {
    float g = gate2[t];
    size_t o = base + (size_t)idx2[t] * CC + p2;
    combine[o]  = g;
    dispatch[o] = (g != 0.0f) ? 1.0f : 0.0f;
  }
}

// ---------------------------------------------------------------------------
// Kernel 6: scalar losses
// balance = mean_{b,e}(proxy_mean * density) * E^2 ; z = lse_sum / B
// ---------------------------------------------------------------------------
__global__ void finalize_kernel(const float* __restrict__ proxy,
                                const int* __restrict__ cnt1,
                                const float* __restrict__ lse_sum,
                                float* __restrict__ out_scalars) {
  if (threadIdx.x == 0 && blockIdx.x == 0) {
    float s = 0.0f;
    for (int i = 0; i < BB * EE; ++i)
      s += (proxy[i] * (1.0f / (float)NN)) * ((float)cnt1[i] * (1.0f / (float)NN));
    out_scalars[0] = s * ((float)(EE * EE) / (float)(BB * EE));  // * 16
    out_scalars[1] = lse_sum[0] * (1.0f / (float)BB);
  }
}

// ---------------------------------------------------------------------------
extern "C" void kernel_launch(void* const* d_in, const int* in_sizes, int n_in,
                              void* d_out, int out_size, void* d_ws, size_t ws_size,
                              hipStream_t stream) {
  const float* x = (const float*)d_in[0];      // [B,N,D] f32
  const float* w = (const float*)d_in[1];      // [D,E]  f32
  float* out = (float*)d_out;

  // workspace carve-up (~527 KB)
  char* ws = (char*)d_ws;
  _Float16* wfrag = (_Float16*)(ws + 0);              // 128 KB
  int*   idx1  = (int*)  (ws + 131072);               // 64 KB
  int*   idx2  = (int*)  (ws + 196608);
  int*   pos1  = (int*)  (ws + 262144);
  int*   pos2  = (int*)  (ws + 327680);
  float* gate1 = (float*)(ws + 393216);
  float* gate2 = (float*)(ws + 458752);
  float* proxy = (float*)(ws + 524288);               // 1 KB
  int*   cnt1  = (int*)  (ws + 525312);               // 1 KB
  float* lse   = (float*)(ws + 526336);               // 4 B

  float* dispatch = out;
  float* combine  = out + SZ;
  float* scalars  = out + 2 * SZ;

  wprep_kernel<<<256, 256, 0, stream>>>(w, wfrag);
  init_kernel<<<1, 256, 0, stream>>>(proxy, cnt1, lse);
  gemm_top2_kernel<<<TT / 64, 128, 0, stream>>>(x, wfrag, idx1, idx2, gate1, gate2, proxy, lse);
  scan_kernel<<<BB, 64, 0, stream>>>(idx1, idx2, pos1, pos2, cnt1);
  zero_kernel<<<8192, 256, 0, stream>>>((v4f*)out, (long long)(2 * SZ / 4));
  scatter_kernel<<<TT / 256, 256, 0, stream>>>(idx1, idx2, pos1, pos2, gate1, gate2, dispatch, combine);
  finalize_kernel<<<1, 1, 0, stream>>>(proxy, cnt1, lse, scalars);
}
